// ClusterLayer_69681549410758
// MI455X (gfx1250) — matrix operand order repaired and checked by reading, not compiled, not run on hardware
//
#include <hip/hip_runtime.h>
#include <hip/hip_bf16.h>

typedef __bf16 bf16_t;
typedef __attribute__((ext_vector_type(16))) __bf16 v16bf;
typedef __attribute__((ext_vector_type(8)))  float  v8f;

#define N_PTS   200000
#define DDIM    256
#define KCLUST  256
#define NTILES  ((N_PTS + 15) / 16)      // 12500 16-point tiles
#define WPB     8                        // waves per block
#define BLOCK   256

// ---------------------------------------------------------------------------
// Prep: centers [K][D] f32  ->  CT [D][K] bf16 (transposed) + csq[K] f32
// ---------------------------------------------------------------------------
__global__ void prep_centers(const float* __restrict__ c,
                             bf16_t* __restrict__ ct,
                             float* __restrict__ csq) {
    __shared__ float red[256];
    const int k = blockIdx.x;    // center row
    const int d = threadIdx.x;   // feature
    float v = c[k * DDIM + d];
    ct[d * KCLUST + k] = (bf16_t)v;          // transposed bf16 store
    red[d] = v * v;
    __syncthreads();
    for (int s = 128; s > 0; s >>= 1) {
        if (d < s) red[d] += red[d + s];
        __syncthreads();
    }
    if (d == 0) csq[k] = red[0];
}

// ---------------------------------------------------------------------------
// Main: each wave computes a 16-point x 256-cluster block of q.
// cross = Z * C^T via v_wmma_f32_16x16x32_bf16; epilogue in registers.
// ---------------------------------------------------------------------------
__global__ void __launch_bounds__(BLOCK, 1)
cluster_q(const float* __restrict__ z,
          const bf16_t* __restrict__ ct,     // [D][K] bf16
          const float* __restrict__ csq,     // [K]
          float* __restrict__ out) {
    // One 32-deep x 256-wide bf16 chunk of CT, shared by the 8 waves (16 KB)
    __shared__ __align__(32) bf16_t lds_ct[32 * KCLUST];

    const int tid  = threadIdx.x;
    const int lane = tid & 31;
    const int wave = tid >> 5;
    const int half = lane >> 4;   // 0: lanes 0-15, 1: lanes 16-31
    const int m16  = lane & 15;

    const int  wtile = blockIdx.x * WPB + wave;
    const bool live  = (wtile < NTILES);
    const int  tile  = live ? wtile : (NTILES - 1);   // clamp: keep EXEC full

    const float* zrow = z + (size_t)(tile * 16 + m16) * DDIM;

    v8f acc[16] = {};          // 16 cluster-tiles x (16x16 f32 / 32 lanes)
    float zsq_part = 0.0f;     // exact fp32 ||z||^2 partial (this lane's K's)

    for (int kk = 0; kk < 8; ++kk) {
        __syncthreads();
        // stage CT rows [kk*32, kk*32+32) x 256 cols  (contiguous 16 KB)
        {
            const uint4* src = (const uint4*)(ct + (size_t)kk * 32 * KCLUST);
            uint4*       dst = (uint4*)lds_ct;
#pragma unroll
            for (int i = 0; i < 4; ++i)
                dst[tid + i * 256] = src[tid + i * 256];
        }
        __syncthreads();

        // ---- A fragment: 16-bit A 16x32 layout (ISA 7.12.2)
        //  lanes 0-15:  K = 0..7 (v0..7), 16..23 (v8..15)
        //  lanes 16-31: K = 8..15        , 24..31
        const int kb = kk * 32 + half * 8;
        float4 f0 = *(const float4*)(zrow + kb);
        float4 f1 = *(const float4*)(zrow + kb + 4);
        float4 f2 = *(const float4*)(zrow + kb + 16);
        float4 f3 = *(const float4*)(zrow + kb + 20);

        zsq_part += f0.x*f0.x + f0.y*f0.y + f0.z*f0.z + f0.w*f0.w
                  + f1.x*f1.x + f1.y*f1.y + f1.z*f1.z + f1.w*f1.w
                  + f2.x*f2.x + f2.y*f2.y + f2.z*f2.z + f2.w*f2.w
                  + f3.x*f3.x + f3.y*f3.y + f3.z*f3.z + f3.w*f3.w;

        v16bf a;
        a[0]  = (bf16_t)f0.x; a[1]  = (bf16_t)f0.y;
        a[2]  = (bf16_t)f0.z; a[3]  = (bf16_t)f0.w;
        a[4]  = (bf16_t)f1.x; a[5]  = (bf16_t)f1.y;
        a[6]  = (bf16_t)f1.z; a[7]  = (bf16_t)f1.w;
        a[8]  = (bf16_t)f2.x; a[9]  = (bf16_t)f2.y;
        a[10] = (bf16_t)f2.z; a[11] = (bf16_t)f2.w;
        a[12] = (bf16_t)f3.x; a[13] = (bf16_t)f3.y;
        a[14] = (bf16_t)f3.z; a[15] = (bf16_t)f3.w;

        // ---- B fragments from LDS: lane holds column N=lane&15, K in the
        // same order as A (8*half + 0..7, 16 + 8*half + 0..7).
#pragma unroll
        for (int t = 0; t < 16; ++t) {
            v16bf b;
#pragma unroll
            for (int s = 0; s < 16; ++s)
                b[s] = lds_ct[(half * 8 + (s & 7) + (s >> 3) * 16) * KCLUST + t * 16 + m16];
            acc[t] = __builtin_amdgcn_wmma_f32_16x16x32_bf16(
                false, a, false, b, (short)0, acc[t], false, false);
        }
    }

    // ---- epilogue -----------------------------------------------------------
    // full ||z||^2 for row (lane&15): this lane's K-half + partner half
    float zsq_full = zsq_part + __shfl_xor(zsq_part, 16);

    float csq_l[16];
#pragma unroll
    for (int t = 0; t < 16; ++t) csq_l[t] = csq[t * 16 + m16];

#pragma unroll
    for (int v = 0; v < 8; ++v) {
        const int m = v + 8 * half;              // output row within tile
        const float zs = __shfl(zsq_full, m);    // lane m holds row m's sum

        float qv[16];
        float s = 0.0f;
#pragma unroll
        for (int t = 0; t < 16; ++t) {
            float cross = acc[t][v];
            float dist  = zs + csq_l[t] - 2.0f * cross;
            dist = fmaxf(dist, 0.0f);
            float q = 1.0f / (1.0f + dist);      // ALPHA=1 -> exponent 1
            qv[t] = q;
            s += q;
        }
        // row sum across the 16 lanes sharing row m (both halves independent)
        s += __shfl_xor(s, 1);
        s += __shfl_xor(s, 2);
        s += __shfl_xor(s, 4);
        s += __shfl_xor(s, 8);
        const float inv = 1.0f / s;

        if (live) {
            float* orow = out + (size_t)(tile * 16 + m) * KCLUST + m16;
#pragma unroll
            for (int t = 0; t < 16; ++t)
                orow[t * 16] = qv[t] * inv;
        }
    }
}

// ---------------------------------------------------------------------------
extern "C" void kernel_launch(void* const* d_in, const int* in_sizes, int n_in,
                              void* d_out, int out_size, void* d_ws, size_t ws_size,
                              hipStream_t stream) {
    const float* z       = (const float*)d_in[0];   // [N, D] f32
    const float* centers = (const float*)d_in[1];   // [K, D] f32
    float*       out     = (float*)d_out;           // [N, K] f32

    bf16_t* ct  = (bf16_t*)d_ws;                                   // 128 KB
    float*  csq = (float*)((char*)d_ws + (size_t)DDIM * KCLUST * 2); // 1 KB

    prep_centers<<<KCLUST, DDIM, 0, stream>>>(centers, ct, csq);

    const int nblocks = (NTILES + WPB - 1) / WPB;   // 1563
    cluster_q<<<nblocks, BLOCK, 0, stream>>>(z, ct, csq, out);
}
